// VIN_63290638074198
// MI455X (gfx1250) — matrix operand order repaired and checked by reading, not compiled, not run on hardware
//
#include <hip/hip_runtime.h>
#include <hip/hip_bf16.h>

typedef __attribute__((ext_vector_type(16))) _Float16 v16h;
typedef __attribute__((ext_vector_type(8)))  float    v8f;

#define VSZ   (66 * 66)        // padded v image (zero halo)
#define ZOFF  VSZ              // 135+ f16 of permanent zeros (hi-lane B source)
#define DOFF  (VSZ + 144)      // 32 f16 dump slots (hi-lane dead stores)
#define VTOT  (VSZ + 176)

// One block per image. 1024 threads = 32 wave32; each wave owns 8 tiles of
// 16 pixels (2 rows of the 64x64 image). Whole VI loop runs in LDS.
__global__ __launch_bounds__(1024, 1)
void VIN_63290638074198_kernel(const float* __restrict__ X,     // [B,2,64,64]
                               const float* __restrict__ h_w,   // [150,2,3,3]
                               const float* __restrict__ h_b,   // [150]
                               const float* __restrict__ r_w,   // [1,150,1,1]
                               const float* __restrict__ q_w,   // [10,1,3,3]
                               const float* __restrict__ w,     // [10,1,3,3]
                               const int*   __restrict__ kp,
                               float* __restrict__ out)         // [B,1,64,64]
{
    __shared__ float    r_pad[VSZ];      // r with 1-px zero halo (f32)
    __shared__ _Float16 vbuf0[VTOT];     // v ping + zero-pad + dump
    __shared__ _Float16 vbuf1[VTOT];     // v pong + zero-pad + dump
    __shared__ float    qw_s[90];        // q_w flat
    __shared__ float    weff_s[19];      // 18 collapsed weights + bias

    const int b    = blockIdx.x;
    const int tid  = threadIdx.x;
    const int lane = tid & 31;
    const int wave = tid >> 5;
    const bool hi  = lane >= 16;
    const int n    = lane & 15;
    const int k    = *kp;

    // ---- Phase 1: init LDS, collapse h_w/r_w/h_b into 2x3x3 eff. conv ----
    for (int i = tid; i < VTOT; i += 1024) {
        if (i < VSZ) r_pad[i] = 0.0f;
        vbuf0[i] = (_Float16)0.0f;
        vbuf1[i] = (_Float16)0.0f;
    }
    if (tid < 90) qw_s[tid] = q_w[tid];
    if (tid < 19) {
        float s = 0.0f;
        if (tid < 18) {
            const int c = tid / 9, pos = tid % 9;
            for (int l = 0; l < 150; ++l) s += r_w[l] * h_w[(l * 2 + c) * 9 + pos];
        } else {
            for (int l = 0; l < 150; ++l) s += r_w[l] * h_b[l];
        }
        weff_s[tid] = s;
    }
    __syncthreads();

    // ---- Phase 2: r = Conv3x3(X, W_eff) + b_eff  (4 pixels / thread) ----
    const float* Xb = X + (size_t)b * 2 * 4096;
    for (int i = 0; i < 4; ++i) {
        const int p = tid + 1024 * i;
        const int y = p >> 6, x = p & 63;
        float s = weff_s[18];
        #pragma unroll
        for (int c = 0; c < 2; ++c)
            #pragma unroll
            for (int dy = 0; dy < 3; ++dy)
                #pragma unroll
                for (int dx = 0; dx < 3; ++dx) {
                    const int yy = y + dy - 1, xx = x + dx - 1;
                    if (yy >= 0 && yy < 64 && xx >= 0 && xx < 64)
                        s += weff_s[c * 9 + dy * 3 + dx] * Xb[c * 4096 + yy * 64 + xx];
                }
        r_pad[(y + 1) * 66 + (x + 1)] = s;
    }
    __syncthreads();

    // ---- Phase 3: build resident WMMA operands ----
    // A: w-weights as 16x32 f16 (M=action, K=tap). Contiguous K layout:
    //   lanes 0-15 : halves 0..8 = taps 0..8 ; rest zero
    //   lanes 16-31: all zero
    v16h Aw = {};
    {
        const int a = n;
        if (a < 10) {
            if (!hi) {
                #pragma unroll
                for (int j = 0; j < 9; ++j) Aw[j] = (_Float16)w[a * 9 + j];
            }
        }
    }

    // Per-tile LDS offsets, with the hi/lo select folded in once:
    // hi lanes read B taps from the permanent-zero region and write their
    // (dead) results to the dump region -> hot loop is fully straight-line.
    int roff[8], woff[8];
    #pragma unroll
    for (int t = 0; t < 8; ++t) {
        const int tau = wave * 8 + t;
        const int y = tau >> 2, x0 = (tau & 3) << 4;
        roff[t] = hi ? ZOFF : (y * 66 + x0 + n);
        woff[t] = hi ? (DOFF + lane) : ((y + 1) * 66 + x0 + n + 1);
    }

    // C tiles: qr[a, pixel] = q_w ⊛ r  (f32, kept in VGPRs for all k iters).
    // D/C layout: lane = pixel-in-tile (mod 16); lanes0-15 rows M=0..7
    // (actions 0-7), lanes16-31 rows M=8..15 (actions 8,9 valid).
    v8f Ct[8];
    #pragma unroll
    for (int t = 0; t < 8; ++t) {
        const int tau = wave * 8 + t;
        const int y = tau >> 2, x0 = (tau & 3) << 4;
        const int px = x0 + n;
        float rt[9];
        #pragma unroll
        for (int tt = 0; tt < 9; ++tt)
            rt[tt] = r_pad[(y + tt / 3) * 66 + px + (tt % 3)];
        v8f c = {};
        const int nch   = hi ? 2 : 8;
        const int abase = hi ? 8 : 0;
        #pragma unroll
        for (int i = 0; i < 8; ++i) {
            float s = 0.0f;
            if (i < nch) {
                #pragma unroll
                for (int tt = 0; tt < 9; ++tt)
                    s += qw_s[(abase + i) * 9 + tt] * rt[tt];
            }
            c[i] = s;
        }
        Ct[t] = c;
    }
    __syncthreads();

    // ---- Phase 4: k value-iteration steps, one WMMA per 16-pixel tile ----
    // v starts at 0, so iteration 0 reproduces v = max_a(q_w ⊛ r).
    // Software-pipelined: B(t+1) is built between WMMA(t) and its D-read,
    // filling the 4-coexec WMMA->VALU hazard window with useful work.
    for (int j = 0; j < k; ++j) {
        const _Float16* cur = (j & 1) ? vbuf1 : vbuf0;
        _Float16*       nxt = (j & 1) ? vbuf0 : vbuf1;

        // B: im2col(v) as 32x16 f16. Contiguous K layout (as A): lanes0-15 hold K=0..15
        // in halves 0..15 (taps 0..8 used); lanes16-31 (K=16..31) all zero
        // (sourced from the zero region).
        v16h Bc = {};
        {
            const _Float16* base = cur + roff[0];
            #pragma unroll
            for (int tt = 0; tt < 9; ++tt) Bc[tt] = base[(tt / 3) * 66 + (tt % 3)];
        }

        #pragma unroll
        for (int t = 0; t < 8; ++t) {
            // q(tile) = w ⊛ v  +  qr(tile)   [single matrix core op]
            v8f d = __builtin_amdgcn_wmma_f32_16x16x32_f16(
                false, Aw, false, Bc, (short)0, Ct[t], false, false);

            if (t < 7) {                       // prefetch next tile's B
                v16h Bn = {};
                const _Float16* base = cur + roff[t + 1];
                #pragma unroll
                for (int tt = 0; tt < 9; ++tt) Bn[tt] = base[(tt / 3) * 66 + (tt % 3)];
                Bc = Bn;
            }

            // v = max over 10 actions, branch-free on the D fragment
            const float m01 = fmaxf(d[0], d[1]);          // actions {0,1} / {8,9}
            const float m23 = fmaxf(d[2], d[3]);
            const float m45 = fmaxf(d[4], d[5]);
            const float m67 = fmaxf(d[6], d[7]);
            const float mlo = fmaxf(fmaxf(m01, m23), fmaxf(m45, m67));
            float m = hi ? m01 : mlo;
            m = fmaxf(m, __shfl_xor(m, 16, 32));
            nxt[woff[t]] = (_Float16)m;                   // hi lanes -> dump
        }
        __syncthreads();
    }

    // ---- Phase 5: write v (f16 -> f32) ----
    const _Float16* fin = (k & 1) ? vbuf1 : vbuf0;
    float* ob = out + (size_t)b * 4096;
    for (int i = 0; i < 4; ++i) {
        const int p = tid + 1024 * i;
        const int y = p >> 6, x = p & 63;
        ob[p] = (float)fin[(y + 1) * 66 + (x + 1)];
    }
}

extern "C" void kernel_launch(void* const* d_in, const int* in_sizes, int n_in,
                              void* d_out, int out_size, void* d_ws, size_t ws_size,
                              hipStream_t stream) {
    (void)n_in; (void)out_size; (void)d_ws; (void)ws_size;
    const float* X   = (const float*)d_in[0];
    const float* h_w = (const float*)d_in[1];
    const float* h_b = (const float*)d_in[2];
    const float* r_w = (const float*)d_in[3];
    const float* q_w = (const float*)d_in[4];
    const float* w   = (const float*)d_in[5];
    const int*   kp  = (const int*)d_in[6];
    float* out = (float*)d_out;

    const int B = in_sizes[0] / (2 * 64 * 64);   // 128
    VIN_63290638074198_kernel<<<B, 1024, 0, stream>>>(X, h_w, h_b, r_w, q_w, w, kp, out);
}
